// Router_10333691314727
// MI455X (gfx1250) — matrix operand order repaired and checked by reading, not compiled, not run on hardware
//
#include <hip/hip_runtime.h>
#include <hip/hip_bf16.h>

typedef __attribute__((ext_vector_type(16))) __bf16 v16bf;
typedef __attribute__((ext_vector_type(8)))  __bf16 v8bf;
typedef __attribute__((ext_vector_type(8)))  float  v8f;

#define TOKENS        16384
#define HIDDEN        4096
#define NEXP          64
#define TOPK          8
#define KC            128                 // K elements per LDS chunk
#define NCHUNK        (HIDDEN / KC)       // 32
#define BROWS         128                 // 64 gate rows + 64 cls rows
#define BSTRIDE       136                 // padded bf16 elems per LDS row (272 B)
#define SLAB          (BROWS * BSTRIDE)   // 17408 bf16 elems
#define SLAB_BYTES    (SLAB * 2)          // 34816 B
#define TOK_PER_BLOCK 256
#define THREADS       256
#define SSTRIDE       69                  // padded fp32 score stride
#define LDS_BYTES     (TOK_PER_BLOCK * SSTRIDE * 4)   // 70656 >= 2*SLAB_BYTES

static __device__ __forceinline__ v16bf cat8(v8bf lo, v8bf hi) {
  return __builtin_shufflevector(lo, hi, 0,1,2,3,4,5,6,7,8,9,10,11,12,13,14,15);
}

// gfx1250 async copy global -> LDS (ASYNCcnt path; decoupled from LOADcnt).
// vdst = wave-relative LDS byte address, voff = 32-bit byte offset, saddr = 64-bit base.
static __device__ __forceinline__ void async_ld_b128(unsigned ldsaddr, unsigned voff,
                                                     const void* base) {
  asm volatile("global_load_async_to_lds_b128 %0, %1, %2"
               :: "v"(ldsaddr), "v"(voff), "s"(base)
               : "memory");
}

__global__ __launch_bounds__(THREADS)
void router_fused(const __bf16* __restrict__ X,
                  const __bf16* __restrict__ Wg,
                  const __bf16* __restrict__ Wc,
                  __hip_bfloat16* __restrict__ wout,
                  int* __restrict__ iout)
{
  extern __shared__ char lds_raw[];
  __bf16* ldsB = (__bf16*)lds_raw;   // phase 1: two weight slabs (async-filled)
  float*  ldsS = (float*)lds_raw;    // phase 2: per-token scores (aliased)

  const int tid   = threadIdx.x;
  const int wave  = tid >> 5;
  const int lane  = tid & 31;
  const int laneN = lane & 15;       // N column / A row within tile
  const int laneH = lane >> 4;       // half-wave selects K sub-block
  const int tokBase = blockIdx.x * TOK_PER_BLOCK;
  const int rsub = tid >> 4;         // 0..15: row within 16-row group
  const int csub = tid & 15;         // 0..15: 16B chunk within row

  // Wave-relative LDS byte offset of the slab base (flat->LDS: low 32 bits).
  const unsigned ldsBase0 = (unsigned)(size_t)(void*)ldsB;

  // ---- async B loader: 8 x b128 per thread per chunk, on ASYNCcnt ----
  unsigned voff[8];   // byte offset from Wg/Wc for next prefetch chunk
  unsigned vlds[8];   // LDS byte address of next prefetch target
  #pragma unroll
  for (int p = 0; p < 8; ++p) {
    const int row = p * 16 + rsub;                     // 0..127 (0-63 gate, 64-127 cls)
    voff[p] = (unsigned)((((row & 63) * HIDDEN) + csub * 8) * 2);
    vlds[p] = ldsBase0 + (unsigned)((row * BSTRIDE + csub * 8) * 2);
  }

  // prologue: chunk 0 -> buffer 0
  #pragma unroll
  for (int p = 0; p < 8; ++p)
    async_ld_b128(vlds[p], voff[p], (p < 4) ? (const void*)Wg : (const void*)Wc);
  int ldsDelta = SLAB_BYTES;         // next target: buffer 1
  #pragma unroll
  for (int p = 0; p < 8; ++p) { voff[p] += KC * 2; vlds[p] += ldsDelta; }
  ldsDelta = -ldsDelta;

  // A row pointers: 16-bit A layout, M = lane&15, two M-tiles per wave
  const __bf16* xrow0 = X + (size_t)(tokBase + wave * 32 + laneN) * HIDDEN;
  const __bf16* xrow1 = xrow0 + (size_t)16 * HIDDEN;

  v8f acc[2][8];                     // [Mtile][0..3 gate | 4..7 cls]
  {
    v8f z = {0.f,0.f,0.f,0.f,0.f,0.f,0.f,0.f};
    #pragma unroll
    for (int mt = 0; mt < 2; ++mt)
      #pragma unroll
      for (int nt = 0; nt < 8; ++nt) acc[mt][nt] = z;
  }

  for (int ch = 0; ch < NCHUNK; ++ch) {
    if (ch + 1 < NCHUNK) {
      // issue chunk ch+1 into the idle buffer; overlaps with this chunk's WMMAs
      #pragma unroll
      for (int p = 0; p < 8; ++p)
        async_ld_b128(vlds[p], voff[p], (p < 4) ? (const void*)Wg : (const void*)Wc);
      #pragma unroll
      for (int p = 0; p < 8; ++p) { voff[p] += KC * 2; vlds[p] += ldsDelta; }
      ldsDelta = -ldsDelta;
      // chunk ch's 8 loads done; the 8 just issued may remain in flight
      asm volatile("s_wait_asynccnt 0x8" ::: "memory");
    } else {
      asm volatile("s_wait_asynccnt 0x0" ::: "memory");
    }
    __syncthreads();

    const __bf16* bslab = ldsB + (ch & 1) * SLAB;

    // A fragments for all 4 K-steps of this chunk (16-bit A layout)
    v16bf a0s[4], a1s[4];
    #pragma unroll
    for (int s = 0; s < 4; ++s) {
      const int ka = ch * KC + s * 32 + laneH * 8;
      a0s[s] = cat8(__builtin_nontemporal_load((const v8bf*)(xrow0 + ka)),
                    __builtin_nontemporal_load((const v8bf*)(xrow0 + ka + 16)));
      a1s[s] = cat8(__builtin_nontemporal_load((const v8bf*)(xrow1 + ka)),
                    __builtin_nontemporal_load((const v8bf*)(xrow1 + ka + 16)));
    }

    // B fragments: explicit one-ahead rotation so the ds_load pair for the
    // next fragment is in flight under the current pair of WMMAs.
    const __bf16* bbase = bslab + laneN * BSTRIDE + laneH * 8;
    #define LOADB(s_, nt_) cat8(*(const v8bf*)(bbase + (nt_) * 16 * BSTRIDE + (s_) * 32), \
                                *(const v8bf*)(bbase + (nt_) * 16 * BSTRIDE + (s_) * 32 + 16))
    v16bf bcur = LOADB(0, 0);
    #pragma unroll
    for (int idx = 0; idx < 32; ++idx) {
      const int s  = idx >> 3;
      const int nt = idx & 7;
      v16bf bnxt = bcur;
      if (idx + 1 < 32) bnxt = LOADB((idx + 1) >> 3, (idx + 1) & 7);
      acc[0][nt] = __builtin_amdgcn_wmma_f32_16x16x32_bf16(
          false, a0s[s], false, bcur, (short)0, acc[0][nt], false, false);
      acc[1][nt] = __builtin_amdgcn_wmma_f32_16x16x32_bf16(
          false, a1s[s], false, bcur, (short)0, acc[1][nt], false, false);
      bcur = bnxt;
    }
    #undef LOADB

    __syncthreads();
  }

  // ---- scores: |c * silu(g)|, bf16-rounded matmul outputs like the reference ----
  #pragma unroll
  for (int mt = 0; mt < 2; ++mt) {
    #pragma unroll
    for (int nt = 0; nt < 4; ++nt) {
      #pragma unroll
      for (int r = 0; r < 8; ++r) {
        float g = __bfloat162float(__float2bfloat16(acc[mt][nt][r]));
        float c = __bfloat162float(__float2bfloat16(acc[mt][nt + 4][r]));
        float sig = 1.0f / (1.0f + __expf(-g));
        float sc  = fabsf(c * g * sig);
        const int tl = wave * 32 + mt * 16 + laneH * 8 + r;   // C/D layout: M = r + 8*halfwave
        ldsS[tl * SSTRIDE + nt * 16 + laneN] = sc;
      }
    }
  }
  __syncthreads();

  // ---- per-token softmax + top-8 (one thread per token) ----
  {
    const int tl = tid;
    float sv[NEXP];
    #pragma unroll
    for (int e = 0; e < NEXP; ++e) sv[e] = ldsS[tl * SSTRIDE + e];

    float m = sv[0];
    #pragma unroll
    for (int e = 1; e < NEXP; ++e) m = fmaxf(m, sv[e]);
    float sum = 0.0f;
    #pragma unroll
    for (int e = 0; e < NEXP; ++e) sum += __expf(sv[e] - m);
    const float inv = 1.0f / sum;

    const size_t gt = (size_t)(tokBase + tl);
    unsigned long long chosen = 0ull;
    for (int j = 0; j < TOPK; ++j) {
      float bv = -3.402823466e+38f;
      int   bi = 0;
      #pragma unroll
      for (int e = 0; e < NEXP; ++e) {
        const bool avail = ((chosen >> e) & 1ull) == 0ull;
        if (avail && sv[e] > bv) { bv = sv[e]; bi = e; }   // strict > => lowest index on ties
      }
      chosen |= (1ull << bi);
      wout[gt * TOPK + j] = __float2bfloat16(__expf(bv - m) * inv);
      iout[gt * TOPK + j] = bi;
    }
  }
}

extern "C" void kernel_launch(void* const* d_in, const int* in_sizes, int n_in,
                              void* d_out, int out_size, void* d_ws, size_t ws_size,
                              hipStream_t stream) {
  (void)in_sizes; (void)n_in; (void)d_ws; (void)ws_size; (void)out_size;
  const __bf16* x  = reinterpret_cast<const __bf16*>(d_in[0]);
  const __bf16* wg = reinterpret_cast<const __bf16*>(d_in[1]);
  const __bf16* wc = reinterpret_cast<const __bf16*>(d_in[2]);
  __hip_bfloat16* wout = reinterpret_cast<__hip_bfloat16*>(d_out);
  int* iout = reinterpret_cast<int*>(wout + (size_t)TOKENS * TOPK);

  static_assert(LDS_BYTES >= 2 * SLAB_BYTES, "LDS union too small");
  hipFuncSetAttribute(reinterpret_cast<const void*>(router_fused),
                      hipFuncAttributeMaxDynamicSharedMemorySize, LDS_BYTES);

  dim3 grid(TOKENS / TOK_PER_BLOCK);   // 64 workgroups
  dim3 block(THREADS);                 // 8 wave32 waves
  router_fused<<<grid, block, LDS_BYTES, stream>>>(x, wg, wc, wout, iout);
}
